// MultiheadAttentionMechanism_38044820308115
// MI455X (gfx1250) — matrix operand order repaired and checked by reading, not compiled, not run on hardware
//
#include <hip/hip_runtime.h>
#include <hip/hip_bf16.h>

typedef __attribute__((ext_vector_type(16))) __bf16 bf16x16;
typedef __attribute__((ext_vector_type(8)))  __bf16 bf16x8;
typedef __attribute__((ext_vector_type(8)))  float  f32x8;

#define B_   16
#define T_   2000
#define E_   512
#define D_   512
#define A_   512
#define H_   4
#define C_   10
#define KW_  201
#define HA_  2048          // H_*A_ (GEMM N dim)
#define KAUG 576           // 512 enc + 40 conv_feat + 24 zero-pad (GEMM K dim)
#define KT_  18            // KAUG / 32  (bf16 WMMA k-steps)
#define ROWS (B_*T_)
#define TT_  125           // T_/16 row tiles per batch

// ---------------------------------------------------------------------------
// Prep 1: pack augmented weight W' [KAUG x HA_] into WMMA-B friendly layout:
//   Wp[kt*65536 + n*32 + klocal] = W'[kt*32+klocal, n]
// so each lane's 16-element B fragment is one contiguous 32B load.
// Rows 0..511   : W_enc[h, k, a]          (n = h*512 + a)
// Rows 512..551 : block-diag W_conv[h,c,a] (row 512 + h*10 + c)
// Rows 552..575 : 0
// ---------------------------------------------------------------------------
__global__ void prep_w_kernel(const float* __restrict__ W_enc,
                              const float* __restrict__ W_conv,
                              __bf16* __restrict__ Wp) {
  int idx = blockIdx.x * 256 + threadIdx.x;
  if (idx >= KT_ * 32 * HA_) return;               // 1,179,648
  int kt  = idx >> 16;
  int rem = idx & 65535;
  int n   = rem >> 5;
  int kl  = rem & 31;
  int k   = kt * 32 + kl;
  int h   = n >> 9;
  int a   = n & 511;
  float v = 0.0f;
  if (k < 512) {
    v = W_enc[((size_t)(h << 9) + k) * 512 + a];
  } else if (k < 552) {
    int j = k - 512;
    int hh = j / 10, c = j % 10;
    if (hh == h) v = W_conv[(size_t)(h * 10 + c) * 512 + a];
  }
  Wp[idx] = (__bf16)v;                             // identity index by construction
}

// ---------------------------------------------------------------------------
// Prep 2a: enc_out fp32 -> bf16 into augmented X cols [0,512)
// ---------------------------------------------------------------------------
__global__ void prep_xenc_kernel(const float* __restrict__ enc,
                                 __bf16* __restrict__ X) {
  int idx = blockIdx.x * 256 + threadIdx.x;
  if (idx >= ROWS * E_) return;                    // 16,384,000
  int row = idx >> 9;
  int e   = idx & 511;
  X[(size_t)row * KAUG + e] = (__bf16)enc[idx];
}

// ---------------------------------------------------------------------------
// Prep 2b: location conv -> X cols [512,552); zero cols [552,576)
// conv_feat[b,t,h,c] = sum_k aw_step[b, t+k-100, h] * conv_w[h*10+c, 0, k]
// ---------------------------------------------------------------------------
__global__ void prep_xconv_kernel(const float* __restrict__ aw_step,
                                  const float* __restrict__ conv_w,
                                  __bf16* __restrict__ X) {
  int idx = blockIdx.x * 256 + threadIdx.x;
  if (idx >= ROWS * 64) return;                    // 2,048,000
  int row = idx >> 6;
  int j   = idx & 63;
  float v = 0.0f;
  if (j < 40) {
    int h = j / 10, c = j % 10;
    int b = row / T_, t = row % T_;
    const float* wk = conv_w + (size_t)(h * 10 + c) * KW_;
    int tau0 = t - 100;
    int k0 = (tau0 < 0) ? -tau0 : 0;
    int k1 = (tau0 + KW_ > T_) ? (T_ - tau0) : KW_;
    const float* ap = aw_step + ((size_t)b * T_ + tau0) * H_ + h;
    float s = 0.0f;
    for (int k = k0; k < k1; ++k) s += ap[(size_t)k * H_] * wk[k];
    v = s;
  }
  X[(size_t)row * KAUG + 512 + j] = (__bf16)v;
}

// ---------------------------------------------------------------------------
// Prep 3: addvec[b, h*512+a] = dec_a[b,h,a] + b_enc[h,a]   (tiny, fp32)
// ---------------------------------------------------------------------------
__global__ void prep_addvec_kernel(const float* __restrict__ dec_out,
                                   const float* __restrict__ W_dec,
                                   const float* __restrict__ b_enc,
                                   float* __restrict__ addvec) {
  int idx = blockIdx.x * 256 + threadIdx.x;
  if (idx >= B_ * HA_) return;                     // 32,768
  int b = idx >> 11;
  int n = idx & 2047;
  int h = n >> 9;
  const float* dp = dec_out + (size_t)b * D_;
  const float* wp = W_dec + (size_t)h * D_ * A_ + (n & 511);
  float s = b_enc[n];
  for (int d = 0; d < D_; ++d) s += dp[d] * wp[(size_t)d * A_];
  addvec[idx] = s;
}

// ---------------------------------------------------------------------------
// Main fused kernel: WMMA GEMM (X[16x576] @ W'[576x2048]) + epilogue
// energy[b,t,h] = mask * sum_a V[h,a]*tanh(enc_a+conv_a+dec_a+b_enc)
// grid: 2000 = B_ * 125 row-tiles. 8 waves; wave w owns cols [w*256, w*256+256).
// ---------------------------------------------------------------------------
__global__ void __launch_bounds__(256) energy_kernel(
    const __bf16* __restrict__ X, const __bf16* __restrict__ Wp,
    const float* __restrict__ addvec, const float* __restrict__ V,
    const int* __restrict__ x_lens, float* __restrict__ energy) {
  __shared__ __align__(16) __bf16 sX[16 * KAUG];   // 18 KB row tile
  __shared__ float eacc[64];                       // [row(16)][h(4)]

  const int tid = threadIdx.x;
  const int b   = blockIdx.x / TT_;
  const int t0  = (blockIdx.x % TT_) * 16;

  if (tid < 64) eacc[tid] = 0.0f;
  // stage 16 contiguous X rows into LDS (16B vectors)
  {
    const uint4* src = (const uint4*)(X + ((size_t)b * T_ + t0) * KAUG);
    uint4* dst = (uint4*)sX;
    for (int i = tid; i < 16 * KAUG * 2 / 16; i += 256) dst[i] = src[i];
  }
  __syncthreads();

  const int wv   = tid >> 5;        // wave 0..7
  const int lane = tid & 31;
  const int m    = lane & 15;       // A row / B column within tile
  const int kh   = lane >> 4;       // k-half select

  f32x8 acc[16];
  #pragma unroll
  for (int ct = 0; ct < 16; ++ct)
    #pragma unroll
    for (int r = 0; r < 8; ++r) acc[ct][r] = 0.0f;

  const __bf16* wbase = Wp + (size_t)(wv * 256 + m) * 32 + kh * 16;

  for (int kt = 0; kt < KT_; ++kt) {
    // A fragment (16x32 bf16) shared across the wave's 16 column tiles
    const __bf16* arow = sX + m * KAUG + kt * 32 + kh * 16;
    bf16x8 a0 = *(const bf16x8*)arow;
    bf16x8 a1 = *(const bf16x8*)(arow + 8);
    bf16x16 af;
    #pragma unroll
    for (int j = 0; j < 8; ++j) { af[j] = a0[j]; af[j + 8] = a1[j]; }

    const __bf16* wkt = wbase + (size_t)kt * (32 * HA_);
    if (kt + 1 < KT_)
      __builtin_prefetch((const void*)(wbase + (size_t)(kt + 1) * (32 * HA_)), 0, 1);

    #pragma unroll
    for (int ct = 0; ct < 16; ++ct) {
      const __bf16* p = wkt + ct * 512;            // next 16 columns
      bf16x8 b0 = *(const bf16x8*)p;
      bf16x8 b1 = *(const bf16x8*)(p + 8);
      bf16x16 bfm;
      #pragma unroll
      for (int j = 0; j < 8; ++j) { bfm[j] = b0[j]; bfm[j + 8] = b1[j]; }
      acc[ct] = __builtin_amdgcn_wmma_f32_16x16x32_bf16(
          false, af, false, bfm, (short)0, acc[ct], false, false);
    }
  }

  // Epilogue: + (dec_a + b_enc), tanh, * V, reduce over this wave's 256 cols.
  float rowp[8];
  #pragma unroll
  for (int r = 0; r < 8; ++r) rowp[r] = 0.0f;
  const int h = wv >> 1;                           // wave's head
  #pragma unroll
  for (int ct = 0; ct < 16; ++ct) {
    int col = wv * 256 + ct * 16 + m;
    float av = addvec[(size_t)b * HA_ + col];
    float vv = V[col];
    #pragma unroll
    for (int r = 0; r < 8; ++r) rowp[r] += tanhf(acc[ct][r] + av) * vv;
  }
  // butterfly reduce across the 16 lanes sharing a row-half
  #pragma unroll
  for (int msk = 1; msk < 16; msk <<= 1)
    #pragma unroll
    for (int r = 0; r < 8; ++r) rowp[r] += __shfl_xor(rowp[r], msk, 32);
  if (m == 0) {
    int rbase = kh * 8;                            // lanes 0/16 hold rows 0-7 / 8-15
    #pragma unroll
    for (int r = 0; r < 8; ++r) atomicAdd(&eacc[(rbase + r) * 4 + h], rowp[r]);
  }
  __syncthreads();

  if (tid < 64) {
    int row = tid >> 2, hh = tid & 3;
    int t = t0 + row;
    float e = (t < x_lens[b]) ? eacc[tid] : 0.0f;  // mask (reference zeroes, no -inf)
    energy[((size_t)b * T_ + t) * H_ + hh] = e;
  }
}

// ---------------------------------------------------------------------------
// Softmax over T per (b,h); padded positions carry energy==0 (matches ref).
// ---------------------------------------------------------------------------
__global__ void softmax_kernel(const float* __restrict__ energy,
                               float* __restrict__ aw) {
  __shared__ float red[256];
  int bi = blockIdx.x;                              // 64 = B_*H_
  int b = bi >> 2, h = bi & 3;
  const float* ep = energy + (size_t)b * T_ * H_ + h;
  int tid = threadIdx.x;
  float mx = -3.4e38f;
  for (int t = tid; t < T_; t += 256) mx = fmaxf(mx, ep[(size_t)t * H_]);
  red[tid] = mx; __syncthreads();
  for (int s = 128; s > 0; s >>= 1) {
    if (tid < s) red[tid] = fmaxf(red[tid], red[tid + s]);
    __syncthreads();
  }
  float M = red[0]; __syncthreads();
  float sum = 0.0f;
  for (int t = tid; t < T_; t += 256) sum += expf(ep[(size_t)t * H_] - M);
  red[tid] = sum; __syncthreads();
  for (int s = 128; s > 0; s >>= 1) {
    if (tid < s) red[tid] += red[tid + s];
    __syncthreads();
  }
  float inv = 1.0f / red[0];
  float* op = aw + (size_t)b * T_ * H_ + h;
  for (int t = tid; t < T_; t += 256) op[(size_t)t * H_] = expf(ep[(size_t)t * H_] - M) * inv;
}

// ---------------------------------------------------------------------------
// ctxcat[b, h*512+e] = sum_t aw[b,t,h] * enc_out[b,t,e]   (fp32, 131 MFLOP)
// ---------------------------------------------------------------------------
__global__ void ctx_kernel(const float* __restrict__ aw,
                           const float* __restrict__ enc,
                           float* __restrict__ ctxcat) {
  int idx = blockIdx.x * 256 + threadIdx.x;
  if (idx >= B_ * H_ * E_) return;                  // 32,768
  int b = idx >> 11;
  int r = idx & 2047;
  int h = r >> 9;
  int e = r & 511;
  const float* ap = aw  + (size_t)b * T_ * H_ + h;
  const float* ep = enc + (size_t)b * T_ * E_ + e;
  float s = 0.0f;
  for (int t = 0; t < T_; ++t) s += ap[(size_t)t * H_] * ep[(size_t)t * E_];
  ctxcat[idx] = s;
}

// ---------------------------------------------------------------------------
// out_ctx[b,e] = ctxcat[b,:] @ W_mha[:,e] + b_mha[e]   (fp32, 33 MFLOP)
// ---------------------------------------------------------------------------
__global__ void final_kernel(const float* __restrict__ ctxcat,
                             const float* __restrict__ W_mha,
                             const float* __restrict__ b_mha,
                             float* __restrict__ out) {
  int idx = blockIdx.x * 256 + threadIdx.x;
  if (idx >= B_ * E_) return;                       // 8,192
  int b = idx >> 9;
  int e = idx & 511;
  const float* cp = ctxcat + (size_t)b * HA_;
  float s = b_mha[e];
  for (int j = 0; j < HA_; ++j) s += cp[j] * W_mha[(size_t)j * E_ + e];
  out[idx] = s;
}

// ---------------------------------------------------------------------------
extern "C" void kernel_launch(void* const* d_in, const int* in_sizes, int n_in,
                              void* d_out, int out_size, void* d_ws, size_t ws_size,
                              hipStream_t stream) {
  const float* enc_out = (const float*)d_in[0];
  const int*   x_lens  = (const int*)  d_in[1];
  const float* dec_out = (const float*)d_in[2];
  const float* aw_step = (const float*)d_in[3];
  const float* W_enc   = (const float*)d_in[4];
  const float* b_enc   = (const float*)d_in[5];
  const float* W_dec   = (const float*)d_in[6];
  const float* W_conv  = (const float*)d_in[7];
  const float* conv_w  = (const float*)d_in[8];
  const float* V       = (const float*)d_in[9];
  const float* W_mha   = (const float*)d_in[10];
  const float* b_mha   = (const float*)d_in[11];

  char* ws = (char*)d_ws;
  __bf16* Xbf    = (__bf16*)(ws);              // B*T*576 bf16       = 36,864,000 B
  __bf16* Wp     = (__bf16*)(ws + 36864000);   // 18*65536 bf16      =  2,359,296 B
  float*  addvec = (float*) (ws + 39223296);   // B*2048 f32         =    131,072 B
  float*  energy = (float*) (ws + 39354368);   // B*T*H f32          =    512,000 B
  float*  ctxcat = (float*) (ws + 39866368);   // B*2048 f32         =    131,072 B

  float* out_ctx = (float*)d_out;              // [B,1,E]  = 8192 floats
  float* aw_out  = (float*)d_out + B_ * E_;    // [B,T,H]  = 128000 floats

  prep_w_kernel     <<<(KT_ * 32 * HA_ + 255) / 256, 256, 0, stream>>>(W_enc, W_conv, Wp);
  prep_xenc_kernel  <<<(ROWS * E_ + 255) / 256,      256, 0, stream>>>(enc_out, Xbf);
  prep_xconv_kernel <<<(ROWS * 64 + 255) / 256,      256, 0, stream>>>(aw_step, conv_w, Xbf);
  prep_addvec_kernel<<<(B_ * HA_ + 255) / 256,       256, 0, stream>>>(dec_out, W_dec, b_enc, addvec);
  energy_kernel     <<<B_ * TT_,                     256, 0, stream>>>(Xbf, Wp, addvec, V, x_lens, energy);
  softmax_kernel    <<<B_ * H_,                      256, 0, stream>>>(energy, aw_out);
  ctx_kernel        <<<(B_ * H_ * E_ + 255) / 256,   256, 0, stream>>>(aw_out, enc_out, ctxcat);
  final_kernel      <<<(B_ * E_ + 255) / 256,        256, 0, stream>>>(ctxcat, W_mha, b_mha, out_ctx);
}
